// GraphAttentionLayer_43207370997882
// MI455X (gfx1250) — matrix operand order repaired and checked by reading, not compiled, not run on hardware
//
#include <hip/hip_runtime.h>

typedef __attribute__((ext_vector_type(8)))  float  v8f;
typedef __attribute__((ext_vector_type(16))) __bf16 v16bf;

#define NODE_DIM  128
#define HEADS     4
#define HIDDEN    32
#define NEG_SLOPE 0.2f
#define GAT_EPS   1e-8f

// ---------------------------------------------------------------------------
// C[M x 128] = A[M x 128] @ Bw^T  (Bw is [128 x 128] row-major, torch [out,in])
// fp32 in, bf16 WMMA, fp32 accumulate. One wave computes a 16-row strip.
// Optional residual + bias (for the output projection).
// ---------------------------------------------------------------------------
__global__ void gat_gemm_bf16_wmma(const float* __restrict__ A,
                                   const float* __restrict__ Bw,
                                   float* __restrict__ C,
                                   const float* __restrict__ residual, // nullable
                                   const float* __restrict__ bias,     // nullable
                                   int nStrips)
{
    const int lane  = threadIdx.x & 31;
    const int wave  = threadIdx.x >> 5;
    const int strip = blockIdx.x * 4 + wave;
    if (strip >= nStrips) return;          // wave-uniform: EXEC all-1s for WMMA

    const int m0   = strip * 16;
    const int mrow = lane & 15;
    const int hi   = lane >> 4;            // lane half (K-split)

    const v8f vz = {0.f,0.f,0.f,0.f,0.f,0.f,0.f,0.f};
    v8f acc[8];
#pragma unroll
    for (int j = 0; j < 8; ++j) acc[j] = vz;

#pragma unroll
    for (int kt = 0; kt < 4; ++kt) {       // K = 128 in 4 steps of 32
        const int kbase = kt * 32;

        // A fragment: 16x32 bf16. Lane m (0-15): K in {0..7, 16..23};
        // lane m+16: K in {8..15, 24..31}; 2 packed values per VGPR.
        const float* xrow = A + (size_t)(m0 + mrow) * NODE_DIM + kbase;
        v16bf afrag;
#pragma unroll
        for (int v = 0; v < 8; ++v) {
            const int k = ((v >> 2) << 4) + (hi << 3) + ((v & 3) << 1);
            afrag[2*v]   = (__bf16)xrow[k];
            afrag[2*v+1] = (__bf16)xrow[k+1];
        }

#pragma unroll
        for (int j = 0; j < 8; ++j) {      // 8 N-tiles of 16
            // B fragment: B[k,n] = Bw[n,k]. Lane n (0-15): K in {0..7, 16..23},
            // lane n+16: K in {8..15, 24..31}. Same K order as A.
            const float* wrow = Bw + (size_t)(16*j + mrow) * NODE_DIM + kbase + (hi << 3);
            v16bf bfrag;
#pragma unroll
            for (int v = 0; v < 8; ++v) {
                const int kk = ((v >> 2) << 4) + ((v & 3) << 1);
                bfrag[2*v]   = (__bf16)wrow[kk];
                bfrag[2*v+1] = (__bf16)wrow[kk+1];
            }
            acc[j] = __builtin_amdgcn_wmma_f32_16x16x32_bf16(
                false, afrag, false, bfrag, (short)0, acc[j], false, false);
        }
    }

    // C layout: VGPR r -> M = 8*hi + r, N = 16*j + (lane&15)
#pragma unroll
    for (int j = 0; j < 8; ++j) {
        const int n = 16*j + mrow;
#pragma unroll
        for (int r = 0; r < 8; ++r) {
            const int m = m0 + (hi << 3) + r;
            float val = acc[j][r];
            if (bias)     val += bias[n];
            const size_t idx = (size_t)m * NODE_DIM + n;
            if (residual) val += residual[idx];
            C[idx] = val;
        }
    }
}

// ---------------------------------------------------------------------------
// Per-node attention logits: s_src[n,h] = dot(h[n,h,:], a_src[h,:]) (+dst).
// One wave per node; lane l covers dims [4l,4l+4) -> flat a index 4l.
// ---------------------------------------------------------------------------
__global__ void gat_scores(const float* __restrict__ h,
                           const float* __restrict__ a_src,
                           const float* __restrict__ a_dst,
                           float* __restrict__ s_src,
                           float* __restrict__ s_dst, int N)
{
    const int lane = threadIdx.x & 31;
    const int node = blockIdx.x * 8 + (threadIdx.x >> 5);
    if (node >= N) return;

    const float4 hv = ((const float4*)(h + (size_t)node * NODE_DIM))[lane];
    const float4 as = ((const float4*)a_src)[lane];
    const float4 ad = ((const float4*)a_dst)[lane];
    float ps = hv.x*as.x + hv.y*as.y + hv.z*as.z + hv.w*as.w;
    float pd = hv.x*ad.x + hv.y*ad.y + hv.z*ad.z + hv.w*ad.w;
#pragma unroll
    for (int off = 1; off < 8; off <<= 1) {   // reduce within 8-lane head group
        ps += __shfl_xor(ps, off, 32);
        pd += __shfl_xor(pd, off, 32);
    }
    if ((lane & 7) == 0) {
        const int head = lane >> 3;
        s_src[node*HEADS + head] = ps;
        s_dst[node*HEADS + head] = pd;
    }
}

// ---------------------------------------------------------------------------
// Edge pass 1: e_exp = exp(leaky_relu(s_src[row]+s_dst[col])); e_sum[col] += e_exp
// One thread per edge, 4 heads each. e_sum (800 KB) is L2-resident.
// ---------------------------------------------------------------------------
__global__ void gat_edge_logits(const int* __restrict__ eidx,
                                const float* __restrict__ s_src,
                                const float* __restrict__ s_dst,
                                float* __restrict__ e_exp,
                                float* __restrict__ e_sum, int E)
{
    const int e = blockIdx.x * blockDim.x + threadIdx.x;
    if (e >= E) return;
    const int row = eidx[e];
    const int col = eidx[E + e];
    const float4 ss = ((const float4*)s_src)[row];
    const float4 sd = ((const float4*)s_dst)[col];
    float4 ex; float v;
    v = ss.x + sd.x; v = v > 0.f ? v : NEG_SLOPE * v; ex.x = __expf(v);
    v = ss.y + sd.y; v = v > 0.f ? v : NEG_SLOPE * v; ex.y = __expf(v);
    v = ss.z + sd.z; v = v > 0.f ? v : NEG_SLOPE * v; ex.z = __expf(v);
    v = ss.w + sd.w; v = v > 0.f ? v : NEG_SLOPE * v; ex.w = __expf(v);
    ((float4*)e_exp)[e] = ex;
    atomicAdd(&e_sum[col*HEADS + 0], ex.x);
    atomicAdd(&e_sum[col*HEADS + 1], ex.y);
    atomicAdd(&e_sum[col*HEADS + 2], ex.z);
    atomicAdd(&e_sum[col*HEADS + 3], ex.w);
}

// ---------------------------------------------------------------------------
// Edge pass 2: agg[col] += alpha * h[row]. One wave per edge; lane l handles
// dims [4l,4l+4) (head = l>>3). 512B coalesced gather + 4 f32 atomics/lane,
// both L2-resident (h, agg = 25.6 MB each << 192 MB L2).
// ---------------------------------------------------------------------------
__global__ void gat_aggregate(const int* __restrict__ eidx,
                              const float* __restrict__ h,
                              const float* __restrict__ e_exp,
                              const float* __restrict__ e_sum,
                              float* __restrict__ agg, int E)
{
    const int lane = threadIdx.x & 31;
    const int e = blockIdx.x * 8 + (threadIdx.x >> 5);
    if (e >= E) return;
    const int row  = eidx[e];
    const int col  = eidx[E + e];
    const int head = lane >> 3;
    const float alpha = e_exp[e*HEADS + head] / (e_sum[col*HEADS + head] + GAT_EPS);
    const float4 hv = ((const float4*)(h + (size_t)row * NODE_DIM))[lane];
    float* dst = agg + (size_t)col * NODE_DIM + lane * 4;
    atomicAdd(dst + 0, alpha * hv.x);
    atomicAdd(dst + 1, alpha * hv.y);
    atomicAdd(dst + 2, alpha * hv.z);
    atomicAdd(dst + 3, alpha * hv.w);
}

// ---------------------------------------------------------------------------
extern "C" void kernel_launch(void* const* d_in, const int* in_sizes, int n_in,
                              void* d_out, int out_size, void* d_ws, size_t ws_size,
                              hipStream_t stream)
{
    const float* nf    = (const float*)d_in[0];
    const int*   eidx  = (const int*)  d_in[1];
    const float* W     = (const float*)d_in[2];
    const float* a_src = (const float*)d_in[3];
    const float* a_dst = (const float*)d_in[4];
    const float* out_W = (const float*)d_in[5];
    const float* out_b = (const float*)d_in[6];
    float* out = (float*)d_out;

    const int N = in_sizes[0] / NODE_DIM;
    const int E = in_sizes[1] / 2;

    // Workspace layout (floats): all offsets 16B-aligned.
    float* ws    = (float*)d_ws;
    float* h     = ws;                                 // N*128
    float* agg   = h     + (size_t)N * NODE_DIM;       // N*128
    float* s_src = agg   + (size_t)N * NODE_DIM;       // N*4
    float* s_dst = s_src + (size_t)N * HEADS;          // N*4
    float* e_sum = s_dst + (size_t)N * HEADS;          // N*4
    float* e_exp = e_sum + (size_t)N * HEADS;          // E*4

    // Deterministic re-init every call (graph-replay safe).
    hipMemsetAsync(agg,   0, (size_t)N * NODE_DIM * sizeof(float), stream);
    hipMemsetAsync(e_sum, 0, (size_t)N * HEADS    * sizeof(float), stream);

    const int nStrips = N / 16;                        // 50000 / 16 = 3125
    dim3 gemmBlk(128);                                 // 4 waves, 1 strip each
    dim3 gemmGrd((nStrips + 3) / 4);

    // 1) h = nf @ W^T           (bf16 WMMA)
    gat_gemm_bf16_wmma<<<gemmGrd, gemmBlk, 0, stream>>>(nf, W, h, nullptr, nullptr, nStrips);
    // 2) per-node logits
    gat_scores<<<(N + 7) / 8, 256, 0, stream>>>(h, a_src, a_dst, s_src, s_dst, N);
    // 3) edge exp + denominator
    gat_edge_logits<<<(E + 255) / 256, 256, 0, stream>>>(eidx, s_src, s_dst, e_exp, e_sum, E);
    // 4) weighted scatter aggregation
    gat_aggregate<<<(E + 7) / 8, 256, 0, stream>>>(eidx, h, e_exp, e_sum, agg, E);
    // 5) out = nf + (agg @ out_W^T + out_b)   (bf16 WMMA, fused residual+bias)
    gat_gemm_bf16_wmma<<<gemmGrd, gemmBlk, 0, stream>>>(agg, out_W, out, nf, out_b, nStrips);
}
